// DiTBlockFrameAttention_41008347742934
// MI455X (gfx1250) — matrix operand order repaired and checked by reading, not compiled, not run on hardware
//
#include <hip/hip_runtime.h>

// ---------------------------------------------------------------------------
// DiT block (frame attention) for MI455X / gfx1250.
// Heavy GEMMs run on v_wmma_f32_16x16x32_bf16 (fp32 accumulate).
// Wave tile 32x64 (8 WMMAs per K-step); K/N are template constants so all
// row strides fold into load immediates (2 base pointers); two-stage (f0/f1)
// software pipeline overlaps loads with WMMA execution. __launch_bounds__
// (256,1) unlocks the full VGPR budget so the pipeline stays in registers.
// ---------------------------------------------------------------------------

typedef __attribute__((ext_vector_type(16))) __bf16 v16bf;
typedef __attribute__((ext_vector_type(8)))  __bf16 v8bf;
typedef __attribute__((ext_vector_type(8)))  float  v8f;

#define HID     1152
#define MLPH    4608
#define NTOK    4096
#define BATCH   8
#define ROWS    (BATCH * NTOK)        // 32768
#define MODS_N  (6 * HID)             // 6912
#define ATTN_SCALE 0.117851130197758f // (1152/16)^-0.5
#define LN_EPS  1e-6f

// float -> bf16, round to nearest even
__device__ __forceinline__ unsigned short f2bf(float f) {
    union { float f; unsigned int u; } v; v.f = f;
    unsigned int u = v.u;
    u += 0x7FFFu + ((u >> 16) & 1u);
    return (unsigned short)(u >> 16);
}

__device__ __forceinline__ float gelu_tanh(float x) {
    return 0.5f * x * (1.f + tanhf(0.7978845608f * (x + 0.044715f * x * x * x)));
}

// blockDim.x == 256 reduction
__device__ __forceinline__ float block_reduce256(float v, float* sbuf) {
    int t = threadIdx.x;
    sbuf[t] = v; __syncthreads();
    for (int s = 128; s > 0; s >>= 1) {
        if (t < s) sbuf[t] += sbuf[t + s];
        __syncthreads();
    }
    float r = sbuf[0];
    __syncthreads();
    return r;
}

// ---------------------------------------------------------------------------
// SiLU(c)  (8x1152)
// ---------------------------------------------------------------------------
__global__ void silu_kernel(const float* __restrict__ c, float* __restrict__ out) {
    int i = blockIdx.x * 256 + threadIdx.x;
    if (i < BATCH * HID) {
        float v = c[i];
        out[i] = v / (1.f + __expf(-v));
    }
}

// ---------------------------------------------------------------------------
// mods = silu(c) @ w_ada + b_ada   (8 x 6912), tiny GEMM, scalar path
// ---------------------------------------------------------------------------
__global__ void mods_kernel(const float* __restrict__ sc,
                            const float* __restrict__ w_ada,
                            const float* __restrict__ b_ada,
                            float* __restrict__ mods) {
    int i = blockIdx.x * 256 + threadIdx.x;
    if (i >= BATCH * MODS_N) return;
    int b = i / MODS_N, j = i % MODS_N;
    const float* cr = sc + b * HID;
    float acc = b_ada[j];
    for (int k = 0; k < HID; ++k)
        acc += cr[k] * w_ada[(size_t)k * MODS_N + j];
    mods[i] = acc;
}

// ---------------------------------------------------------------------------
// fp32 (K x N) -> bf16 transposed (N x K), LDS-tiled (coalesced both sides)
// ---------------------------------------------------------------------------
__global__ void transpose_bf16_kernel(const float* __restrict__ src,
                                      unsigned short* __restrict__ dst,
                                      int K, int N) {
    __shared__ float tile[32][33];
    int n0 = blockIdx.x * 32, k0 = blockIdx.y * 32;
    int tx = threadIdx.x, ty = threadIdx.y;
    for (int i = 0; i < 4; ++i)
        tile[ty + 8 * i][tx] = src[(size_t)(k0 + ty + 8 * i) * N + n0 + tx];
    __syncthreads();
    for (int i = 0; i < 4; ++i)
        dst[(size_t)(n0 + ty + 8 * i) * K + k0 + tx] = f2bf(tile[tx][ty + 8 * i]);
}

// ---------------------------------------------------------------------------
// x0 rows: LN + modulate(sh_msa, sc_msa) -> fp32 (16 x 1152)
// ---------------------------------------------------------------------------
__global__ void x0_ln_mod_kernel(const float* __restrict__ x,
                                 const float* __restrict__ mods,
                                 float* __restrict__ x0n) {
    __shared__ float sbuf[256];
    int blk = blockIdx.x;           // 0..15
    int b = blk >> 1, m = blk & 1;
    const float* r = x + ((size_t)b * NTOK + m) * HID;
    float s = 0.f, ss = 0.f;
    for (int c = threadIdx.x; c < HID; c += 256) { float v = r[c]; s += v; ss += v * v; }
    float tot  = block_reduce256(s,  sbuf);
    float tot2 = block_reduce256(ss, sbuf);
    float mu = tot / HID;
    float rstd = rsqrtf(tot2 / HID - mu * mu + LN_EPS);
    const float* sh  = mods + b * MODS_N;         // sh_msa
    const float* scm = mods + b * MODS_N + HID;   // sc_msa
    for (int c = threadIdx.x; c < HID; c += 256) {
        float v = (r[c] - mu) * rstd;
        x0n[(size_t)blk * HID + c] = v * (1.f + scm[c]) + sh[c];
    }
}

// ---------------------------------------------------------------------------
// kv = x0n @ w1 + b1   (16 x 2304), tiny
// ---------------------------------------------------------------------------
__global__ void kv_kernel(const float* __restrict__ x0n,
                          const float* __restrict__ w1,
                          const float* __restrict__ b1,
                          float* __restrict__ kv) {
    int i = blockIdx.x * 256 + threadIdx.x;
    if (i >= 16 * 2304) return;
    int bm = i / 2304, j = i % 2304;
    const float* xr = x0n + (size_t)bm * HID;
    float acc = b1[j];
    for (int k = 0; k < HID; ++k)
        acc += xr[k] * w1[(size_t)k * 2304 + j];
    kv[i] = acc;
}

// ---------------------------------------------------------------------------
// Per-row LN + modulate -> bf16 (one block per row; 32768 rows)
// ---------------------------------------------------------------------------
__global__ void ln_mod_bf16_kernel(const float* __restrict__ src,
                                   const float* __restrict__ mods,
                                   unsigned short* __restrict__ dst,
                                   int shOff, int scOff) {
    __shared__ float sbuf[256];
    int row = blockIdx.x;
    int b = row >> 12;
    const float* r = src + (size_t)row * HID;
    float s = 0.f, ss = 0.f;
    for (int c = threadIdx.x; c < HID; c += 256) { float v = r[c]; s += v; ss += v * v; }
    float tot  = block_reduce256(s,  sbuf);
    float tot2 = block_reduce256(ss, sbuf);
    float mu = tot / HID;
    float rstd = rsqrtf(tot2 / HID - mu * mu + LN_EPS);
    const float* sh = mods + b * MODS_N + shOff;
    const float* sc = mods + b * MODS_N + scOff;
    for (int c = threadIdx.x; c < HID; c += 256) {
        float v = (r[c] - mu) * rstd;
        v = v * (1.f + sc[c]) + sh[c];
        dst[(size_t)row * HID + c] = f2bf(v);
    }
}

// ---------------------------------------------------------------------------
// WMMA bf16 GEMM:  out[M,N] = A[M,K] * B^T[N,K] + bias   (K, N compile-time)
//   block = 256 threads (8 waves, 4(M) x 2(N) wave grid); block tile 128 x 128;
//   each wave: 32 x 64 tile = 2 A-frags x 4 B-frags = 8 WMMAs per K-step.
//   All row strides are immediates off two base pointers; two-stage f0/f1
//   pipeline (K-step 64) overlaps global loads with WMMA execution.
//   __launch_bounds__(256, 1): full VGPR budget, no spills.
// Fragment packing (wave32, 16-bit data, ISA 7.12.2):
//   A 16x32: lane L holds row (L&15); halves = K {koff..koff+16} (contiguous, same as B),
//            koff = (L<16)?0:16  -> two 16B loads.
//   B 32x16 (from B^T row = output column): lane L holds col (L&15);
//            halves = K {boff..boff+16}, boff = (L<16)?0:16 -> one 32B load.
//   C/D 16x16: VGPR r, lane L -> (row = r + 8*(L>=16), col = L&15).
// MODE 0: fp32 store (+bias)        -> q projection
// MODE 1: gelu(+bias) -> bf16 store -> fc1
// MODE 2: resid + g_mlp*(+bias)     -> fc2 / final output
// ---------------------------------------------------------------------------
struct GemmFrags {
    v16bf a0, a1;
    v16bf b[4];
};

template <int K>
__device__ __forceinline__ void gemm_load_frags(GemmFrags& f,
                                                const __bf16* __restrict__ Abase,
                                                const __bf16* __restrict__ Bbase,
                                                int k0) {
    v8bf a0lo = *(const v8bf*)(Abase + k0);
    v8bf a0hi = *(const v8bf*)(Abase + k0 + 8);
    v8bf a1lo = *(const v8bf*)(Abase + 16 * K + k0);
    v8bf a1hi = *(const v8bf*)(Abase + 16 * K + k0 + 8);
    #pragma unroll
    for (int i = 0; i < 8; ++i) {
        f.a0[i] = a0lo[i]; f.a0[i + 8] = a0hi[i];
        f.a1[i] = a1lo[i]; f.a1[i + 8] = a1hi[i];
    }
    #pragma unroll
    for (int t = 0; t < 4; ++t)
        f.b[t] = *(const v16bf*)(Bbase + t * 16 * K + k0);
}

__device__ __forceinline__ void gemm_mma8(v8f acc[2][4], const GemmFrags& f) {
    #pragma unroll
    for (int t = 0; t < 4; ++t) {
        acc[0][t] = __builtin_amdgcn_wmma_f32_16x16x32_bf16(
            false, f.a0, false, f.b[t], (short)0, acc[0][t], false, false);
        acc[1][t] = __builtin_amdgcn_wmma_f32_16x16x32_bf16(
            false, f.a1, false, f.b[t], (short)0, acc[1][t], false, false);
    }
}

template <int MODE, int K, int N>
__global__ __launch_bounds__(256, 1)
void gemm_bf16_kernel(const unsigned short* __restrict__ Au,
                      const unsigned short* __restrict__ Bu,
                      const float* __restrict__ bias,
                      float* __restrict__ outF,
                      unsigned short* __restrict__ outB,
                      const float* __restrict__ resid,
                      const float* __restrict__ mods) {
    const __bf16* A = reinterpret_cast<const __bf16*>(Au);
    const __bf16* B = reinterpret_cast<const __bf16*>(Bu);
    const int lane = threadIdx.x & 31;
    const int wv   = threadIdx.x >> 5;
    const int wm   = wv & 3;            // 4 M-subtiles
    const int wn   = wv >> 2;           // 2 N-subtiles
    const int m0   = blockIdx.x * 128 + wm * 32;
    const int n0   = blockIdx.y * 128 + wn * 64;
    const int l15  = lane & 15;
    const int koff = (lane < 16) ? 0 : 16;
    const int boff = (lane < 16) ? 0 : 16;

    v8f acc[2][4];
    #pragma unroll
    for (int s = 0; s < 2; ++s)
        #pragma unroll
        for (int t = 0; t < 4; ++t)
            #pragma unroll
            for (int i = 0; i < 8; ++i) acc[s][t][i] = 0.f;

    const __bf16* Abase = A + (size_t)(m0 + l15) * K + koff;
    const __bf16* Bbase = B + (size_t)(n0 + l15) * K + boff;

    // two-stage software pipeline over K (step 64); K/32 is even for both K
    constexpr int STEPS = K / 32;
    static_assert(STEPS % 2 == 0, "K/32 must be even");
    GemmFrags f0, f1;
    gemm_load_frags<K>(f0, Abase, Bbase, 0);
    gemm_load_frags<K>(f1, Abase, Bbase, 32);
    #pragma unroll 1
    for (int s = 0; s < STEPS - 2; s += 2) {
        gemm_mma8(acc, f0);
        gemm_load_frags<K>(f0, Abase, Bbase, (s + 2) * 32);
        gemm_mma8(acc, f1);
        gemm_load_frags<K>(f1, Abase, Bbase, (s + 3) * 32);
    }
    gemm_mma8(acc, f0);
    gemm_mma8(acc, f1);

    #pragma unroll
    for (int s = 0; s < 2; ++s) {
        const int rbase = m0 + 16 * s + ((lane >= 16) ? 8 : 0);
        #pragma unroll
        for (int t = 0; t < 4; ++t) {
            int col = n0 + 16 * t + l15;
            float bv = bias[col];
            #pragma unroll
            for (int r = 0; r < 8; ++r) {
                int grow = rbase + r;
                float val = acc[s][t][r] + bv;
                if (MODE == 0) {
                    outF[(size_t)grow * N + col] = val;
                } else if (MODE == 1) {
                    outB[(size_t)grow * N + col] = f2bf(gelu_tanh(val));
                } else {
                    int b = grow >> 12;
                    float gm = mods[b * MODS_N + 5 * HID + col];   // g_mlp
                    size_t oi = (size_t)grow * HID + col;
                    outF[oi] = resid[oi] + gm * val;
                }
            }
        }
    }
}

// ---------------------------------------------------------------------------
// Rank-2 attention: softmax over 2 keys, y = a0*v0 + a1*v1, in-place on Q.
// ---------------------------------------------------------------------------
__global__ void attn_kernel(float* __restrict__ Q, const float* __restrict__ kv) {
    __shared__ float sbuf[256];
    int row = blockIdx.x;
    int b = row >> 12;
    const float* k0 = kv + (size_t)b * 2 * 2304;
    const float* k1 = k0 + 2304;
    const float* v0 = k0 + HID;
    const float* v1 = k1 + HID;
    float* q = Q + (size_t)row * HID;
    float s0 = 0.f, s1 = 0.f;
    for (int c = threadIdx.x; c < HID; c += 256) {
        float qv = q[c];
        s0 += qv * k0[c];
        s1 += qv * k1[c];
    }
    float d0 = block_reduce256(s0, sbuf) * ATTN_SCALE;
    float d1 = block_reduce256(s1, sbuf) * ATTN_SCALE;
    float mx = fmaxf(d0, d1);
    float e0 = __expf(d0 - mx), e1 = __expf(d1 - mx);
    float inv = 1.f / (e0 + e1);
    float a0 = e0 * inv, a1 = e1 * inv;
    for (int c = threadIdx.x; c < HID; c += 256)
        q[c] = a0 * v0[c] + a1 * v1[c];
}

// ---------------------------------------------------------------------------
// swapaxes(1,2).reshape == per-batch flat transpose of (4096 x 1152);
// fused with (1 + g_msa) gate at store. LDS-tiled, coalesced both sides.
// ---------------------------------------------------------------------------
__global__ void permute_gate_kernel(const float* __restrict__ Y,
                                    const float* __restrict__ mods,
                                    float* __restrict__ Yg) {
    __shared__ float tile[32][33];
    int b  = blockIdx.z;
    int c0 = blockIdx.x * 32;   // column (hidden) index of source view
    int n0 = blockIdx.y * 32;   // row (token) index of source view
    const float* Yb = Y  + (size_t)b * NTOK * HID;
    float*       Og = Yg + (size_t)b * NTOK * HID;
    int tx = threadIdx.x, ty = threadIdx.y;
    for (int i = 0; i < 4; ++i)
        tile[ty + 8 * i][tx] = Yb[(size_t)(n0 + ty + 8 * i) * HID + c0 + tx];
    __syncthreads();
    const float* g = mods + b * MODS_N + 2 * HID;   // g_msa
    for (int i = 0; i < 4; ++i) {
        int r = c0 + ty + 8 * i;          // 0..1151
        int s = n0 + tx;                  // 0..4095
        int oidx = r * 4096 + s;          // flat index into (4096 x 1152) output
        int cg = oidx % HID;              // gate column in permuted space
        Og[oidx] = (1.f + g[cg]) * tile[tx][ty + 8 * i];
    }
}

// ---------------------------------------------------------------------------
// Host launch
// ---------------------------------------------------------------------------
extern "C" void kernel_launch(void* const* d_in, const int* in_sizes, int n_in,
                              void* d_out, int out_size, void* d_ws, size_t ws_size,
                              hipStream_t stream) {
    const float* x      = (const float*)d_in[0];
    const float* c      = (const float*)d_in[1];
    const float* w_ada  = (const float*)d_in[2];
    const float* b_ada  = (const float*)d_in[3];
    const float* w1     = (const float*)d_in[4];
    const float* b1     = (const float*)d_in[5];
    const float* w2     = (const float*)d_in[6];
    const float* b2     = (const float*)d_in[7];
    const float* w_fc1  = (const float*)d_in[8];
    const float* b_fc1  = (const float*)d_in[9];
    const float* w_fc2  = (const float*)d_in[10];
    const float* b_fc2  = (const float*)d_in[11];
    float* out = (float*)d_out;

    char* ws = (char*)d_ws;
    size_t off = 0;
    auto alloc = [&](size_t bytes) -> void* {
        void* p = ws + off;
        off += (bytes + 255) & ~(size_t)255;
        return p;
    };
    float*          silu_c = (float*)alloc((size_t)BATCH * HID * 4);
    float*          mods   = (float*)alloc((size_t)BATCH * MODS_N * 4);
    float*          x0n    = (float*)alloc((size_t)16 * HID * 4);
    float*          kv     = (float*)alloc((size_t)16 * 2304 * 4);
    unsigned short* w2T    = (unsigned short*)alloc((size_t)HID * HID * 2);
    unsigned short* wfc1T  = (unsigned short*)alloc((size_t)MLPH * HID * 2);
    unsigned short* wfc2T  = (unsigned short*)alloc((size_t)HID * MLPH * 2);
    unsigned short* Xn     = (unsigned short*)alloc((size_t)ROWS * HID * 2);   // also reused as Hn
    float*          Q      = (float*)alloc((size_t)ROWS * HID * 4);            // q, then y (in place)
    float*          Yg     = (float*)alloc((size_t)ROWS * HID * 4);            // permuted+gated residual
    unsigned short* G      = (unsigned short*)alloc((size_t)ROWS * MLPH * 2);  // gelu(fc1) bf16

    // adaLN conditioning
    silu_kernel<<<(BATCH * HID + 255) / 256, 256, 0, stream>>>(c, silu_c);
    mods_kernel<<<(BATCH * MODS_N + 255) / 256, 256, 0, stream>>>(silu_c, w_ada, b_ada, mods);

    // weights -> bf16 B^T (N x K) layout
    transpose_bf16_kernel<<<dim3(HID / 32, HID / 32),  dim3(32, 8), 0, stream>>>(w2,    w2T,   HID,  HID);
    transpose_bf16_kernel<<<dim3(MLPH / 32, HID / 32), dim3(32, 8), 0, stream>>>(w_fc1, wfc1T, HID,  MLPH);
    transpose_bf16_kernel<<<dim3(HID / 32, MLPH / 32), dim3(32, 8), 0, stream>>>(w_fc2, wfc2T, MLPH, HID);

    // k, v from the two masked frames (tiny)
    x0_ln_mod_kernel<<<16, 256, 0, stream>>>(x, mods, x0n);
    kv_kernel<<<(16 * 2304 + 255) / 256, 256, 0, stream>>>(x0n, w1, b1, kv);

    // q = modulate(ln(x)) @ w2 + b2  (WMMA)
    ln_mod_bf16_kernel<<<ROWS, 256, 0, stream>>>(x, mods, Xn, 0, HID);
    gemm_bf16_kernel<0, HID, HID><<<dim3(ROWS / 128, HID / 128), 256, 0, stream>>>(
        Xn, w2T, b2, Q, nullptr, nullptr, nullptr);

    // rank-2 softmax attention, y written in place of q
    attn_kernel<<<ROWS, 256, 0, stream>>>(Q, kv);

    // swapaxes-reshape permute + (1+g_msa) gate
    permute_gate_kernel<<<dim3(HID / 32, NTOK / 32, BATCH), dim3(32, 8), 0, stream>>>(Q, mods, Yg);

    // mlp: gelu(modulate(ln(Yg)) @ w_fc1 + b_fc1) @ w_fc2 + b_fc2, fused residual
    ln_mod_bf16_kernel<<<ROWS, 256, 0, stream>>>(Yg, mods, Xn, 3 * HID, 4 * HID);
    gemm_bf16_kernel<1, HID, MLPH><<<dim3(ROWS / 128, MLPH / 128), 256, 0, stream>>>(
        Xn, wfc1T, b_fc1, nullptr, G, nullptr, nullptr);
    gemm_bf16_kernel<2, MLPH, HID><<<dim3(ROWS / 128, HID / 128), 256, 0, stream>>>(
        G, wfc2T, b_fc2, out, nullptr, Yg, mods);
}
